// DualPathCodebookPlanRouter_28338194219134
// MI455X (gfx1250) — compile-verified, matrix operand order checked by reading, not run on hardware
//
#include <hip/hip_runtime.h>
#include <math.h>

// ---------------- constants ----------------
#define BB 4
#define TT 256
#define SS 512
#define DD 1024
#define QQN 4
#define HH 512
#define GDD 3072
#define M1 (BB * TT)   // 1024
#define M2 (BB * SS)   // 2048

#define TILE_K 32
#define TILE_M 128
#define TILE_N 128
#define LDS_STRIDE 40  // u16 units -> 80B row stride, keeps 16B alignment

typedef __bf16 bf16_t;
typedef bf16_t v16bf __attribute__((ext_vector_type(16)));
typedef float v8f __attribute__((ext_vector_type(8)));
typedef unsigned short v4u16 __attribute__((ext_vector_type(4)));
typedef unsigned short v8u16 __attribute__((ext_vector_type(8)));
typedef unsigned short v16u16 __attribute__((ext_vector_type(16)));

#if __has_builtin(__builtin_amdgcn_global_load_async_to_lds_b128)
#define ASYNC_LDS 1
typedef int v4i_t __attribute__((ext_vector_type(4)));
typedef __attribute__((address_space(1))) v4i_t* gv4i_p;
typedef __attribute__((address_space(3))) v4i_t* lv4i_p;
#else
#define ASYNC_LDS 0
#endif

// hardware RNE convert (v_cvt_*bf16_f32) instead of a 4-op bit trick
__device__ inline unsigned short f2bf(float f) {
  bf16_t h = (bf16_t)f;
  return __builtin_bit_cast(unsigned short, h);
}
__device__ inline float bf2f(unsigned short h) {
  return __uint_as_float(((unsigned int)h) << 16);
}

// Load one 16x32 bf16 fragment (16 rows, contiguous k) from LDS.
// CDNA5 16-bit A/B layout: lanes 0-15 -> row L, K = {0..7, 16..23};
// lanes 16-31 -> row L-16, K = {8..15, 24..31}.
__device__ inline v16bf lds_frag(const unsigned short* tile, int rbase, int lane) {
  const int half = lane >> 4;
  const unsigned short* p = tile + (rbase + (lane & 15)) * LDS_STRIDE + half * 8;
  v8u16 lo = *(const v8u16*)(p);
  v8u16 hi = *(const v8u16*)(p + 16);
  v16u16 f;
#pragma unroll
  for (int i = 0; i < 8; ++i) { f[i] = lo[i]; f[i + 8] = hi[i]; }
  return __builtin_bit_cast(v16bf, f);
}

// ---------------- generic WMMA GEMM (A bf16, weights f32 transposed on the fly) ----
// C[M,N] = A[M,K] * B([K,N] logical, inner block Ninner for gate_W1 layout) + bias
// MODE: 0 store bf16; 1 qv-mask -> bf16+f32; 2 exact gelu -> bf16;
//       3 route pass1: out = qh + s*lam*v (masked); 4 route pass2: out += s*(1-lam)*v
template <int MODE>
__global__ __launch_bounds__(256) void gemm_wmma_bf16(
    const unsigned short* __restrict__ Ah,
    const float* __restrict__ Bw, int K, int N, int Ninner,
    const float* __restrict__ bias,
    unsigned short* __restrict__ out_bf, float* __restrict__ out_f,
    const float* __restrict__ qvf, const float* __restrict__ lam,
    const float* __restrict__ scale, const float* __restrict__ qh,
    int q_idx) {
  __shared__ unsigned short lds_a[TILE_M * LDS_STRIDE];
  __shared__ unsigned short lds_b[TILE_N * LDS_STRIDE];

  const int tid = threadIdx.x;
  const int lane = tid & 31;
  const int wave = tid >> 5;
  const int wm = (wave & 3) * 32;
  const int wn = (wave >> 2) * 64;
  const int m0 = blockIdx.y * TILE_M;
  const int n0 = blockIdx.x * TILE_N;

  v8f acc[2][4];
#pragma unroll
  for (int mi = 0; mi < 2; ++mi)
#pragma unroll
    for (int ni = 0; ni < 4; ++ni)
#pragma unroll
      for (int i = 0; i < 8; ++i) acc[mi][ni][i] = 0.0f;

  for (int kk = 0; kk < K; kk += TILE_K) {
    // ---- stage A tile (128 x 32 bf16): 512 groups of 8 bf16 (16B), 2 per thread
#pragma unroll
    for (int i = 0; i < 2; ++i) {
      int g = tid + i * 256;
      int r = g >> 2;
      int kg = g & 3;
      const unsigned short* src = Ah + (long)(m0 + r) * K + kk + kg * 8;
      unsigned short* dst = &lds_a[r * LDS_STRIDE + kg * 8];
#if ASYNC_LDS
      __builtin_amdgcn_global_load_async_to_lds_b128((gv4i_p)src, (lv4i_p)dst, 0, 0);
#else
      *(v8u16*)dst = *(const v8u16*)src;
#endif
    }
    // ---- stage B tile transposed: float4 along n (coalesced), b16 column stores
#pragma unroll
    for (int i = 0; i < 4; ++i) {
      int e = tid + i * 256;
      int k = e >> 5;   // 0..31
      int ng = e & 31;  // group of 4 columns
      int gn = n0 + ng * 4;
      int qq = gn / Ninner;
      long idx = ((long)qq * K + (kk + k)) * (long)Ninner + (gn % Ninner);
      const float4 w = *(const float4*)(Bw + idx);
      unsigned short* dst = &lds_b[(ng * 4) * LDS_STRIDE + k];
      dst[0] = f2bf(w.x);
      dst[LDS_STRIDE] = f2bf(w.y);
      dst[2 * LDS_STRIDE] = f2bf(w.z);
      dst[3 * LDS_STRIDE] = f2bf(w.w);
    }
    if (kk + TILE_K < K) {
      int gn = n0 + (tid & 127);
      int qq = gn / Ninner;
      long pidx = ((long)qq * K + (kk + TILE_K + (tid >> 7) * 16)) * (long)Ninner + (gn % Ninner);
      __builtin_prefetch(&Bw[pidx], 0, 1);  // global_prefetch_b8 next weight tile
    }
#if ASYNC_LDS
#if __has_builtin(__builtin_amdgcn_s_wait_asynccnt)
    __builtin_amdgcn_s_wait_asynccnt(0);
#else
    asm volatile("s_wait_asynccnt 0x0" ::: "memory");
#endif
#endif
    __syncthreads();

    v16bf afrag[2], bfrag[4];
#pragma unroll
    for (int mi = 0; mi < 2; ++mi) afrag[mi] = lds_frag(lds_a, wm + mi * 16, lane);
#pragma unroll
    for (int ni = 0; ni < 4; ++ni) bfrag[ni] = lds_frag(lds_b, wn + ni * 16, lane);
#pragma unroll
    for (int mi = 0; mi < 2; ++mi)
#pragma unroll
      for (int ni = 0; ni < 4; ++ni)
        acc[mi][ni] = __builtin_amdgcn_wmma_f32_16x16x32_bf16(
            false, afrag[mi], false, bfrag[ni], (short)0, acc[mi][ni], false, false);
    __syncthreads();
  }

  // epilogue: acc VGPR i -> M = i + 8*(lane>=16); N = lane&15
#pragma unroll
  for (int mi = 0; mi < 2; ++mi) {
#pragma unroll
    for (int ni = 0; ni < 4; ++ni) {
      const int colt = n0 + wn + ni * 16 + (lane & 15);
      const float bcol = bias[colt];
#pragma unroll
      for (int i = 0; i < 8; ++i) {
        const int rowt = m0 + wm + mi * 16 + (lane >> 4) * 8 + i;
        float v = acc[mi][ni][i] + bcol;
        if (MODE == 0) {
          out_bf[(long)rowt * N + colt] = f2bf(v);
        } else if (MODE == 1) {
          float g = (qvf[rowt] > 0.5f) ? v : 0.0f;
          out_f[(long)rowt * N + colt] = g;
          out_bf[(long)rowt * N + colt] = f2bf(g);
        } else if (MODE == 2) {
          float g = 0.5f * v * (1.0f + erff(v * 0.70710678118654752f));
          out_bf[(long)rowt * N + colt] = f2bf(g);
        } else {
          const int brow = rowt / TT;
          const int trow = rowt % TT;
          const long oidx = (((long)brow * QQN + q_idx) * TT + trow) * DD + colt;
          const float s = scale[q_idx];
          const float l = lam[rowt * QQN + q_idx];
          const bool ok = qvf[rowt] > 0.5f;
          if (MODE == 3) {
            out_f[oidx] = qh[(long)rowt * DD + colt] + (ok ? s * l * v : 0.0f);
          } else {
            if (ok) out_f[oidx] += s * (1.0f - l) * v;
          }
        }
      }
    }
  }
}

// ---------------- windowed attention pooling (both paths) ----------------
__global__ __launch_bounds__(256) void attn_pool_kernel(
    const unsigned short* __restrict__ qc, const unsigned short* __restrict__ kc,
    const unsigned short* __restrict__ vc, const unsigned short* __restrict__ qb,
    const unsigned short* __restrict__ kb, const unsigned short* __restrict__ vb,
    const int* __restrict__ qpos, const int* __restrict__ clens,
    unsigned short* __restrict__ cs_pre, unsigned short* __restrict__ bs_pre,
    float* __restrict__ qvf) {
  const int lane = threadIdx.x & 31;
  const int wave = threadIdx.x >> 5;
  const int m = blockIdx.x * 8 + wave;  // [0, 1024)
  const int b = m >> 8;
  const int base = qpos[m];
  const int cl = clens[b];
  const int dbase = lane * 32;  // each lane owns 32 contiguous d

  // hoist q vectors (32 bf16 per path) into registers
  v8u16 qcr[4], qbr[4];
#pragma unroll
  for (int i = 0; i < 4; ++i) {
    qcr[i] = *(const v8u16*)(qc + m * DD + dbase + i * 8);
    qbr[i] = *(const v8u16*)(qb + m * DD + dbase + i * 8);
  }

  float sc[6], sb[6];
  int rows[6];
  bool val[6];
  bool any = false;
#pragma unroll
  for (int j = 0; j < 6; ++j) {
    int p = base + (j - 1);  // offsets -1..4
    bool v = (base >= 0) && (p >= 0) && (p < cl);
    int pc = p < 0 ? 0 : (p > SS - 1 ? SS - 1 : p);
    rows[j] = b * SS + pc;
    val[j] = v;
    any = any || v;
    float dc = 0.0f, db = 0.0f;
    if (v) {
#pragma unroll
      for (int i = 0; i < 4; ++i) {
        v8u16 kcv = *(const v8u16*)(kc + rows[j] * DD + dbase + i * 8);
        v8u16 kbv = *(const v8u16*)(kb + rows[j] * DD + dbase + i * 8);
#pragma unroll
        for (int e = 0; e < 8; ++e) {
          dc += bf2f(qcr[i][e]) * bf2f(kcv[e]);
          db += bf2f(qbr[i][e]) * bf2f(kbv[e]);
        }
      }
    }
    for (int off = 16; off > 0; off >>= 1) {
      dc += __shfl_xor(dc, off, 32);
      db += __shfl_xor(db, off, 32);
    }
    sc[j] = dc * (1.0f / 32.0f);  // / sqrt(1024)
    sb[j] = db * (1.0f / 32.0f);
  }
  float ac[6], ab[6];
  float mc = -1e30f, mb = -1e30f;
#pragma unroll
  for (int j = 0; j < 6; ++j)
    if (val[j]) { mc = fmaxf(mc, sc[j]); mb = fmaxf(mb, sb[j]); }
  float dc = 0.0f, db = 0.0f;
#pragma unroll
  for (int j = 0; j < 6; ++j) {
    ac[j] = val[j] ? __expf(sc[j] - mc) : 0.0f;
    ab[j] = val[j] ? __expf(sb[j] - mb) : 0.0f;
    dc += ac[j];
    db += ab[j];
  }
  dc = fmaxf(dc, 1e-6f);
  db = fmaxf(db, 1e-6f);
#pragma unroll
  for (int j = 0; j < 6; ++j) { ac[j] /= dc; ab[j] /= db; }

#pragma unroll
  for (int i = 0; i < 4; ++i) {
    float oc[8], ob[8];
#pragma unroll
    for (int e = 0; e < 8; ++e) { oc[e] = 0.0f; ob[e] = 0.0f; }
#pragma unroll
    for (int j = 0; j < 6; ++j) {
      if (val[j]) {
        v8u16 vcv = *(const v8u16*)(vc + rows[j] * DD + dbase + i * 8);
        v8u16 vbv = *(const v8u16*)(vb + rows[j] * DD + dbase + i * 8);
#pragma unroll
        for (int e = 0; e < 8; ++e) {
          oc[e] += ac[j] * bf2f(vcv[e]);
          ob[e] += ab[j] * bf2f(vbv[e]);
        }
      }
    }
    v8u16 hc, hb;
#pragma unroll
    for (int e = 0; e < 8; ++e) { hc[e] = f2bf(oc[e]); hb[e] = f2bf(ob[e]); }
    *(v8u16*)(cs_pre + m * DD + dbase + i * 8) = hc;
    *(v8u16*)(bs_pre + m * DD + dbase + i * 8) = hb;
  }
  if (lane == 0) qvf[m] = any ? 1.0f : 0.0f;
}

// ---------------- f32 -> bf16 convert (vectorized x4) ----------------
__global__ __launch_bounds__(256) void cvt_bf16_kernel(const float* __restrict__ in,
                                                       unsigned short* __restrict__ out,
                                                       int n4) {
  int i = blockIdx.x * 256 + threadIdx.x;
  if (i < n4) {
    float4 v = ((const float4*)in)[i];
    v4u16 h;
    h[0] = f2bf(v.x); h[1] = f2bf(v.y); h[2] = f2bf(v.z); h[3] = f2bf(v.w);
    ((v4u16*)out)[i] = h;
  }
}

// ---------------- LayerNorm over concat [qh | cs | bs] ----------------
__global__ __launch_bounds__(256) void layernorm_concat_kernel(
    const float* __restrict__ qh, const float* __restrict__ csf,
    const float* __restrict__ bsf, const float* __restrict__ lnw,
    const float* __restrict__ lnb, unsigned short* __restrict__ gin) {
  __shared__ float xs[GDD];
  __shared__ float red0[8], red1[8];
  const int m = blockIdx.x;
  const int lane = threadIdx.x & 31;
  const int wave = threadIdx.x >> 5;
  float s = 0.0f, s2 = 0.0f;
  for (int g = threadIdx.x; g < GDD; g += 256) {
    float x = (g < DD) ? qh[m * DD + g]
                       : ((g < 2 * DD) ? csf[m * DD + g - DD] : bsf[m * DD + g - 2 * DD]);
    xs[g] = x;
    s += x;
    s2 += x * x;
  }
  for (int off = 16; off > 0; off >>= 1) {
    s += __shfl_xor(s, off, 32);
    s2 += __shfl_xor(s2, off, 32);
  }
  if (lane == 0) { red0[wave] = s; red1[wave] = s2; }
  __syncthreads();
  float ts = 0.0f, t2 = 0.0f;
#pragma unroll
  for (int i = 0; i < 8; ++i) { ts += red0[i]; t2 += red1[i]; }
  const float mu = ts / (float)GDD;
  const float var = t2 / (float)GDD - mu * mu;
  const float rstd = rsqrtf(var + 1e-5f);
  for (int g = threadIdx.x; g < GDD; g += 256)
    gin[m * GDD + g] = f2bf((xs[g] - mu) * rstd * lnw[g] + lnb[g]);
}

// ---------------- lambda = sigmoid(h . W2 + b2) ----------------
__global__ __launch_bounds__(256) void lam_kernel(
    const unsigned short* __restrict__ hbf, const float* __restrict__ W2,
    const float* __restrict__ b2, const float* __restrict__ qvf,
    float* __restrict__ lam_ws, float* __restrict__ dout_lam) {
  const int lane = threadIdx.x & 31;
  const int wave = threadIdx.x >> 5;
  const int w = blockIdx.x * 8 + wave;  // [0, 4096)
  const int m = w >> 2;
  const int q = w & 3;
  float acc = 0.0f;
#pragma unroll
  for (int i = 0; i < 2; ++i) {
    const int hh = lane * 16 + i * 8;  // each lane owns 16 contiguous h
    v8u16 hv = *(const v8u16*)(hbf + m * (QQN * HH) + q * HH + hh);
#pragma unroll
    for (int e = 0; e < 8; ++e) acc += bf2f(hv[e]) * W2[q * HH + hh + e];
  }
  for (int off = 16; off > 0; off >>= 1) acc += __shfl_xor(acc, off, 32);
  if (lane == 0) {
    float l = 1.0f / (1.0f + __expf(-(acc + b2[q])));
    lam_ws[m * QQN + q] = l;
    dout_lam[m * QQN + q] = (qvf[m] > 0.5f) ? l : 0.5f;
  }
}

// ---------------- tail: qv + route_scale sections of d_out ----------------
__global__ __launch_bounds__(256) void tail_kernel(const float* __restrict__ qvf,
                                                   const float* __restrict__ rs,
                                                   float* __restrict__ out_qv,
                                                   float* __restrict__ out_scale) {
  int i = blockIdx.x * 256 + threadIdx.x;
  if (i < M1) out_qv[i] = qvf[i];
  if (i < QQN) out_scale[i] = rs[i];
}

// ---------------- host launcher ----------------
extern "C" void kernel_launch(void* const* d_in, const int* in_sizes, int n_in,
                              void* d_out, int out_size, void* d_ws, size_t ws_size,
                              hipStream_t stream) {
  (void)in_sizes; (void)n_in; (void)out_size; (void)ws_size;
  const float* qh = (const float*)d_in[0];
  const float* ph = (const float*)d_in[1];
  const int* qpos = (const int*)d_in[2];
  const int* clens = (const int*)d_in[3];
  const float* Wq_c = (const float*)d_in[4];  const float* bq_c = (const float*)d_in[5];
  const float* Wk_c = (const float*)d_in[6];  const float* bk_c = (const float*)d_in[7];
  const float* Wv_c = (const float*)d_in[8];  const float* bv_c = (const float*)d_in[9];
  const float* Wq_b = (const float*)d_in[10]; const float* bq_b = (const float*)d_in[11];
  const float* Wk_b = (const float*)d_in[12]; const float* bk_b = (const float*)d_in[13];
  const float* Wv_b = (const float*)d_in[14]; const float* bv_b = (const float*)d_in[15];
  const float* Wo_c = (const float*)d_in[16]; const float* bo_c = (const float*)d_in[17];
  const float* Wo_b = (const float*)d_in[18]; const float* bo_b = (const float*)d_in[19];
  const float* lnw = (const float*)d_in[20];  const float* lnb = (const float*)d_in[21];
  const float* gW1 = (const float*)d_in[22];  const float* gb1 = (const float*)d_in[23];
  const float* gW2 = (const float*)d_in[24];  const float* gb2 = (const float*)d_in[25];
  const float* rWc = (const float*)d_in[26];  const float* rbc = (const float*)d_in[27];
  const float* rWb = (const float*)d_in[28];  const float* rbb = (const float*)d_in[29];
  const float* rscale = (const float*)d_in[30];

  char* ws = (char*)d_ws;
  size_t off = 0;
  auto alloc = [&](size_t bytes) -> void* {
    void* p = ws + off;
    off = (off + bytes + 255) & ~(size_t)255;
    return p;
  };
  unsigned short* qhb = (unsigned short*)alloc((size_t)M1 * DD * 2);
  unsigned short* phb = (unsigned short*)alloc((size_t)M2 * DD * 2);
  unsigned short* qc = (unsigned short*)alloc((size_t)M1 * DD * 2);
  unsigned short* qb = (unsigned short*)alloc((size_t)M1 * DD * 2);
  unsigned short* kc = (unsigned short*)alloc((size_t)M2 * DD * 2);
  unsigned short* vc = (unsigned short*)alloc((size_t)M2 * DD * 2);
  unsigned short* kb = (unsigned short*)alloc((size_t)M2 * DD * 2);
  unsigned short* vb = (unsigned short*)alloc((size_t)M2 * DD * 2);
  unsigned short* csp = (unsigned short*)alloc((size_t)M1 * DD * 2);
  unsigned short* bsp = (unsigned short*)alloc((size_t)M1 * DD * 2);
  float* qvf = (float*)alloc((size_t)M1 * 4);
  unsigned short* csb = (unsigned short*)alloc((size_t)M1 * DD * 2);
  unsigned short* bsb = (unsigned short*)alloc((size_t)M1 * DD * 2);
  float* csf = (float*)alloc((size_t)M1 * DD * 4);
  float* bsf = (float*)alloc((size_t)M1 * DD * 4);
  unsigned short* gin = (unsigned short*)alloc((size_t)M1 * GDD * 2);
  unsigned short* hbf = (unsigned short*)alloc((size_t)M1 * QQN * HH * 2);
  float* lamw = (float*)alloc((size_t)M1 * QQN * 4);

  const long OFF_LAM = (long)BB * QQN * TT * DD;      // 4194304
  const long OFF_QV = OFF_LAM + (long)BB * TT * QQN;  // +4096
  const long OFF_SC = OFF_QV + (long)BB * TT;         // +1024
  float* outf = (float*)d_out;

  // activations -> bf16
  cvt_bf16_kernel<<<(M1 * DD / 4 + 255) / 256, 256, 0, stream>>>(qh, qhb, M1 * DD / 4);
  cvt_bf16_kernel<<<(M2 * DD / 4 + 255) / 256, 256, 0, stream>>>(ph, phb, M2 * DD / 4);

  dim3 blk(256);
  // Q projections
  gemm_wmma_bf16<0><<<dim3(DD / TILE_N, M1 / TILE_M), blk, 0, stream>>>(
      qhb, Wq_c, DD, DD, DD, bq_c, qc, nullptr, nullptr, nullptr, nullptr, nullptr, 0);
  gemm_wmma_bf16<0><<<dim3(DD / TILE_N, M1 / TILE_M), blk, 0, stream>>>(
      qhb, Wq_b, DD, DD, DD, bq_b, qb, nullptr, nullptr, nullptr, nullptr, nullptr, 0);
  // K/V projections on plan_hidden once (gather windows afterwards): 3x less GEMM work
  gemm_wmma_bf16<0><<<dim3(DD / TILE_N, M2 / TILE_M), blk, 0, stream>>>(
      phb, Wk_c, DD, DD, DD, bk_c, kc, nullptr, nullptr, nullptr, nullptr, nullptr, 0);
  gemm_wmma_bf16<0><<<dim3(DD / TILE_N, M2 / TILE_M), blk, 0, stream>>>(
      phb, Wv_c, DD, DD, DD, bv_c, vc, nullptr, nullptr, nullptr, nullptr, nullptr, 0);
  gemm_wmma_bf16<0><<<dim3(DD / TILE_N, M2 / TILE_M), blk, 0, stream>>>(
      phb, Wk_b, DD, DD, DD, bk_b, kb, nullptr, nullptr, nullptr, nullptr, nullptr, 0);
  gemm_wmma_bf16<0><<<dim3(DD / TILE_N, M2 / TILE_M), blk, 0, stream>>>(
      phb, Wv_b, DD, DD, DD, bv_b, vb, nullptr, nullptr, nullptr, nullptr, nullptr, 0);

  // windowed masked-softmax pooling, both paths, + qv
  attn_pool_kernel<<<M1 / 8, blk, 0, stream>>>(qc, kc, vc, qb, kb, vb, qpos, clens,
                                               csp, bsp, qvf);

  // output projections with qv row-mask, dual bf16+f32 outputs
  gemm_wmma_bf16<1><<<dim3(DD / TILE_N, M1 / TILE_M), blk, 0, stream>>>(
      csp, Wo_c, DD, DD, DD, bo_c, csb, csf, qvf, nullptr, nullptr, nullptr, 0);
  gemm_wmma_bf16<1><<<dim3(DD / TILE_N, M1 / TILE_M), blk, 0, stream>>>(
      bsp, Wo_b, DD, DD, DD, bo_b, bsb, bsf, qvf, nullptr, nullptr, nullptr, 0);

  // LayerNorm over 3072-wide concat
  layernorm_concat_kernel<<<M1, blk, 0, stream>>>(qh, csf, bsf, lnw, lnb, gin);

  // gate: [1024 x 3072] x [3072 x 2048] (gate_W1 batched layout via Ninner=512), gelu fused
  gemm_wmma_bf16<2><<<dim3((QQN * HH) / TILE_N, M1 / TILE_M), blk, 0, stream>>>(
      gin, gW1, GDD, QQN * HH, HH, gb1, hbf, nullptr, nullptr, nullptr, nullptr, nullptr, 0);

  // lambda + lam_out section of d_out
  lam_kernel<<<(M1 * QQN) / 8, blk, 0, stream>>>(hbf, gW2, gb2, qvf, lamw, outf + OFF_LAM);

  // route: two accumulate passes per q with fully fused combine epilogue
  for (int q = 0; q < QQN; ++q) {
    gemm_wmma_bf16<3><<<dim3(DD / TILE_N, M1 / TILE_M), blk, 0, stream>>>(
        csb, rWc + (size_t)q * DD * DD, DD, DD, DD, rbc + (size_t)q * DD,
        nullptr, outf, qvf, lamw, rscale, qh, q);
    gemm_wmma_bf16<4><<<dim3(DD / TILE_N, M1 / TILE_M), blk, 0, stream>>>(
        bsb, rWb + (size_t)q * DD * DD, DD, DD, DD, rbb + (size_t)q * DD,
        nullptr, outf, qvf, lamw, rscale, qh, q);
  }

  // qv + route_scale sections of d_out
  tail_kernel<<<4, blk, 0, stream>>>(qvf, rscale, outf + OFF_QV, outf + OFF_SC);
}
